// FFTDeepfakeDetector_57432302682256
// MI455X (gfx1250) — matrix operand order repaired and checked by reading, not compiled, not run on hardware
//
#include <hip/hip_runtime.h>
#include <hip/hip_bf16.h>
#include <math.h>

typedef _Float16 half_t;
typedef __attribute__((ext_vector_type(16))) _Float16 v16h;
typedef __attribute__((ext_vector_type(8)))  _Float16 v8h;
typedef __attribute__((ext_vector_type(8)))  float    v8f;
typedef __attribute__((ext_vector_type(2)))  float    v2f;

#define IMG 224
#define NPX (IMG*IMG)          // 50176
#define TWO_PI_OVER_N (6.2831853071795864769f/224.0f)

__device__ inline v8f vz8f(){ v8f v; 
#pragma unroll
  for(int i=0;i<8;++i) v[i]=0.f; return v; }
__device__ inline v8h vz8h(){ v8h v;
#pragma unroll
  for(int i=0;i<8;++i) v[i]=(_Float16)0.f; return v; }
__device__ inline v16h cat8(v8h lo, v8h hi){ v16h r;
#pragma unroll
  for(int i=0;i<8;++i){ r[i]=lo[i]; r[i+8]=hi[i]; } return r; }

__device__ inline v8f wmma16(v16h a, v16h b, v8f c){
  return __builtin_amdgcn_wmma_f32_16x16x32_f16(false, a, false, b, (short)0, c, false, false);
}
__device__ inline v8f wmma4(v2f a, v2f b, v8f c){
  return __builtin_amdgcn_wmma_f32_16x16x4_f32(false, a, false, b, (short)0, c, false, false);
}
__device__ inline float silu_f(float x){ return x / (1.f + __expf(-x)); }

// ---------------------------------------------------------------- utility
__global__ __launch_bounds__(256) void k_zero(float* p, int n){
  int i = blockIdx.x*256 + threadIdx.x;
  if(i<n) p[i]=0.f;
}

// Build DFT cos/sin matrices directly in WMMA f32 16x16x4 A-fragment layout.
// A-frag map (f32, 16x4): lane<16 -> m=lane, k={0,1}; lane>=16 -> m=lane-16, k={2,3}
__global__ __launch_bounds__(256) void k_build_dft(float* cf, float* sf){
  int tid = blockIdx.x*256 + threadIdx.x;          // 14*56*64 = 50176 entries
  if(tid >= 14*56*64) return;
  int e    = tid & 1;
  int lane = (tid>>1) & 31;
  int fk   = tid >> 6;
  int kt   = fk % 56;
  int mt   = fk / 56;
  int u = mt*16 + (lane&15);
  int y = kt*4 + ((lane&16)?2:0) + e;
  int ang = (u*y) % IMG;
  float th = (float)ang * TWO_PI_OVER_N;
  cf[tid] =  __cosf(th);
  sf[tid] = -__sinf(th);
}

// ---------------------------------------------------------------- grayscale
__global__ __launch_bounds__(256) void k_gray(const float* __restrict__ x, float* __restrict__ gray){
  int b = blockIdx.y;
  int p = blockIdx.x*256 + threadIdx.x;
  if(p >= NPX) return;
  size_t base = (size_t)b*3*NPX;
  gray[(size_t)b*NPX + p] = 0.299f*x[base+p] + 0.587f*x[base+NPX+p] + 0.114f*x[base+2*NPX+p];
}

// ---------------------------------------------------------------- DFT stage 1: U = (F*G)^T, f32 WMMA
__global__ __launch_bounds__(32) void k_dft1(const float* __restrict__ gray,
                                             const float* __restrict__ cf, const float* __restrict__ sf,
                                             float* __restrict__ ure, float* __restrict__ uim){
  const int b  = blockIdx.y;
  const int mt = blockIdx.x / 14;   // u tile
  const int nt = blockIdx.x % 14;   // x tile
  const int lane = threadIdx.x;
  const int n0 = lane & 15;
  const int off2 = (lane&16) ? 2 : 0;
  const int mb   = (lane&16) ? 8 : 0;
  const float* g = gray + (size_t)b*NPX;
  const int x = nt*16 + n0;
  v8f ar = vz8f(), ai = vz8f();
  for(int kt=0; kt<56; ++kt){
    int k0 = kt*4 + off2;
    v2f bf; bf[0] = g[k0*IMG + x]; bf[1] = g[(k0+1)*IMG + x];
    v2f ac = *(const v2f*)(cf + (mt*56+kt)*64 + lane*2);
    v2f as = *(const v2f*)(sf + (mt*56+kt)*64 + lane*2);
    ar = wmma4(ac, bf, ar);
    ai = wmma4(as, bf, ai);
  }
  // store transposed: U[x][u]; per lane 8 consecutive u values
  size_t base = (size_t)b*NPX + (size_t)x*IMG + mt*16 + mb;
  *(v8f*)(ure + base) = ar;
  *(v8f*)(uim + base) = ai;
}

// ---------------------------------------------------------------- DFT stage 2: S^T = F*U, |.|, log1p, fftshift
__global__ __launch_bounds__(32) void k_dft2(const float* __restrict__ ure, const float* __restrict__ uim,
                                             const float* __restrict__ cf, const float* __restrict__ sf,
                                             float* __restrict__ lm){
  const int b  = blockIdx.y;
  const int mt = blockIdx.x / 14;   // v tile
  const int nt = blockIdx.x % 14;   // u tile
  const int lane = threadIdx.x;
  const int n0 = lane & 15;
  const int off2 = (lane&16) ? 2 : 0;
  const int mb   = (lane&16) ? 8 : 0;
  const float* pr = ure + (size_t)b*NPX;
  const float* pi = uim + (size_t)b*NPX;
  const int u = nt*16 + n0;
  v8f cr = vz8f(), si = vz8f(), ci = vz8f(), sr = vz8f();
  for(int kt=0; kt<56; ++kt){
    int k0 = kt*4 + off2;
    v2f br; br[0] = pr[k0*IMG + u]; br[1] = pr[(k0+1)*IMG + u];
    v2f bi; bi[0] = pi[k0*IMG + u]; bi[1] = pi[(k0+1)*IMG + u];
    v2f ac = *(const v2f*)(cf + (mt*56+kt)*64 + lane*2);
    v2f as = *(const v2f*)(sf + (mt*56+kt)*64 + lane*2);
    cr = wmma4(ac, br, cr);
    si = wmma4(as, bi, si);
    ci = wmma4(ac, bi, ci);
    sr = wmma4(as, br, sr);
  }
  v8f out;
#pragma unroll
  for(int r=0;r<8;++r){
    float re = cr[r] - si[r];
    float im = ci[r] + sr[r];
    out[r] = log1pf(sqrtf(re*re + im*im));
  }
  int vbase = mt*16 + mb;
  int us = u + 112;     if(us >= IMG) us -= IMG;
  int vs = vbase + 112; if(vs >= IMG) vs -= IMG;
  *(v8f*)(lm + (size_t)b*NPX + (size_t)us*IMG + vs) = out;
}

// ---------------------------------------------------------------- per-sample mean / inv-std (ddof=1)
__global__ __launch_bounds__(256) void k_stats2d(const float* __restrict__ lm, float* __restrict__ stats){
  int b = blockIdx.x, tid = threadIdx.x;
  __shared__ float s1[256], s2[256];
  float a=0.f, q=0.f;
  for(int i=tid;i<NPX;i+=256){ float v = lm[(size_t)b*NPX+i]; a+=v; q+=v*v; }
  s1[tid]=a; s2[tid]=q; __syncthreads();
  for(int st=128; st>0; st>>=1){ if(tid<st){ s1[tid]+=s1[tid+st]; s2[tid]+=s2[tid+st]; } __syncthreads(); }
  if(tid==0){
    const float n = (float)NPX;
    float mean = s1[0]/n;
    float var  = fmaxf((s2[0] - s1[0]*s1[0]/n)/(n-1.f), 0.f);
    stats[b*2+0] = mean;
    stats[b*2+1] = 1.f/(sqrtf(var)+1e-8f);
  }
}

// ---------------------------------------------------------------- radial bins
__device__ inline int radial_id(int i){
  int y = i/IMG, x = i%IMG;
  float dy = (float)(y-112), dx = (float)(x-112);
  int r = (int)rintf(sqrtf(dy*dy+dx*dx));
  return (r<112) ? r : 112;
}
__global__ __launch_bounds__(256) void k_radial_counts(float* counts){
  __shared__ float sH[113];
  int tid = threadIdx.x;
  if(tid<113) sH[tid]=0.f;
  __syncthreads();
  for(int i=tid;i<NPX;i+=256) atomicAdd(&sH[radial_id(i)], 1.f);
  __syncthreads();
  if(tid<113) counts[tid]=sH[tid];
}
__global__ __launch_bounds__(256) void k_radial_sums(const float* __restrict__ lm, float* __restrict__ rsum){
  int b = blockIdx.x, tid = threadIdx.x;
  __shared__ float sH[113];
  if(tid<113) sH[tid]=0.f;
  __syncthreads();
  for(int i=tid;i<NPX;i+=256) atomicAdd(&sH[radial_id(i)], lm[(size_t)b*NPX+i]);
  __syncthreads();
  if(tid<113) rsum[b*113+tid]=sH[tid];
}
__global__ __launch_bounds__(128) void k_radial_norm(const float* __restrict__ rsum, const float* __restrict__ counts,
                                                     half_t* __restrict__ rad){
  int b = blockIdx.x, i = threadIdx.x;   // 128 threads, 112 real bins, pad K to 128
  __shared__ float s1[128], s2[128];
  float v = 0.f;
  if(i<112) v = rsum[b*113+i] / (counts[i]+1e-8f);
  s1[i] = (i<112)? v : 0.f;
  s2[i] = (i<112)? v*v : 0.f;
  __syncthreads();
  for(int st=64; st>0; st>>=1){ if(i<st){ s1[i]+=s1[i+st]; s2[i]+=s2[i+st]; } __syncthreads(); }
  float mean = s1[0]/112.f;
  float var  = fmaxf((s2[0] - s1[0]*s1[0]/112.f)/111.f, 0.f);
  float sd   = sqrtf(var)+1e-8f;
  rad[b*128+i] = (half_t)((i<112)? (v-mean)/sd : 0.f);
}

// ---------------------------------------------------------------- BN fold: scale=g/sqrt(v+eps); shift=cb*scale + b - m*scale
__global__ __launch_bounds__(256) void k_fold(const float* g, const float* bb, const float* m, const float* v,
                                              const float* cb, float* scale, float* shift, int C){
  int i = threadIdx.x + blockIdx.x*256;
  if(i>=C) return;
  float inv = g[i]*rsqrtf(v[i]+1e-5f);
  scale[i] = inv;
  shift[i] = cb[i]*inv + bb[i] - m[i]*inv;
}

// ---------------------------------------------------------------- weight repack -> WMMA f16 A-fragment layout
// A-frag map (f16, 16x32): m = lane&15 ; k = (lane&16?16:0) + e
__global__ __launch_bounds__(256) void k_pack_wconv(const float* __restrict__ w, const float* __restrict__ scale,
                                                    half_t* __restrict__ out, int Cout, int Cin){
  int KC = Cin/32, MT = Cout/16;
  int N = MT*9*KC*512;
  int tid = blockIdx.x*256 + threadIdx.x;
  if(tid>=N) return;
  int e    = tid & 15;
  int lane = (tid>>4) & 31;
  int f    = tid >> 9;
  int kc   = f % KC;
  int tap  = (f/KC) % 9;
  int mt   = f/(KC*9);
  int co = mt*16 + (lane&15);
  int kk = ((lane&16)?16:0) + e;
  int ci = kc*32 + kk;
  out[tid] = (half_t)(w[(co*Cin + ci)*9 + tap] * scale[co]);
}

// Linear weights [Nf][Kreal] -> B-fragment layout, zero-pad to Kpad, optional per-n scale.
// B-frag map (f16, 32x16): n = lane&15 ; k = (e&8?16:0) + (lane&16?8:0) + (e&7)
__global__ __launch_bounds__(256) void k_pack_lin(const float* __restrict__ w, const float* __restrict__ scale,
                                                  half_t* __restrict__ out, int Nf, int Kreal, int Kpad){
  int NT = Nf/16;
  int total = (Kpad/32)*NT*512;
  int tid = blockIdx.x*256 + threadIdx.x;
  if(tid>=total) return;
  int e    = tid & 15;
  int lane = (tid>>4) & 31;
  int f    = tid >> 9;
  int nt   = f % NT;
  int kc   = f / NT;
  int n = nt*16 + (lane&15);
  int k = kc*32 + ((e&8)?16:0) + ((lane&16)?8:0) + (e&7);
  float v = 0.f;
  if(k < Kreal) v = w[n*Kreal + k] * (scale ? scale[n] : 1.f);
  out[tid] = (half_t)v;
}

// ---------------------------------------------------------------- conv1 (Cin=1): direct VALU + BN + SiLU + maxpool2
__global__ __launch_bounds__(256) void k_conv1(const float* __restrict__ lm, const float* __restrict__ stats,
                                               const float* __restrict__ w1, const float* __restrict__ scale1,
                                               const float* __restrict__ shift1, half_t* __restrict__ out){
  int b = blockIdx.y, patch = blockIdx.x;
  int PY = patch/7, PX = patch%7;
  int tid = threadIdx.x;
  __shared__ float sSpec[34*34];
  __shared__ float sW[288];
  __shared__ float sSc[32], sSh[32];
  float mean = stats[b*2], invs = stats[b*2+1];
  int iy0 = PY*32 - 1, ix0 = PX*32 - 1;
  for(int i=tid;i<34*34;i+=256){
    int hy=i/34, hx=i%34;
    int iy=iy0+hy, ix=ix0+hx;
    sSpec[i] = ((unsigned)iy<224u && (unsigned)ix<224u) ? (lm[(size_t)b*NPX + iy*IMG + ix]-mean)*invs : 0.f;
  }
  if(tid<288) sW[tid]=w1[tid];
  if(tid<32){ sSc[tid]=scale1[tid]; sSh[tid]=shift1[tid]; }
  __syncthreads();
  int oy = tid>>4, ox = tid&15;     // pooled pixel within 16x16 patch
  float p[4][4];
#pragma unroll
  for(int yy=0;yy<4;++yy)
#pragma unroll
    for(int xx=0;xx<4;++xx) p[yy][xx] = sSpec[(2*oy+yy)*34 + (2*ox+xx)];
  size_t obase = (((size_t)(b*112 + PY*16+oy))*112 + (PX*16+ox))*32;
  for(int co=0;co<32;++co){
    const float* wc = sW + co*9;
    float mx = -1e30f;
#pragma unroll
    for(int sy=0;sy<2;++sy)
#pragma unroll
      for(int sx=0;sx<2;++sx){
        float s = 0.f;
#pragma unroll
        for(int dy=0;dy<3;++dy)
#pragma unroll
          for(int dx=0;dx<3;++dx) s += wc[dy*3+dx]*p[sy+dy][sx+dx];
        mx = fmaxf(mx, silu_f(sSc[co]*s + sSh[co]));
      }
    out[obase + co] = (half_t)mx;
  }
}

// ---------------------------------------------------------------- conv2: implicit GEMM, WMMA f16, + pool
__global__ __launch_bounds__(256) void k_conv2(const half_t* __restrict__ in, const half_t* __restrict__ wf,
                                               const float* __restrict__ shift, half_t* __restrict__ out){
  int b = blockIdx.y, patch = blockIdx.x;
  int PY = patch/7, PX = patch%7;
  int tid = threadIdx.x, lane = tid&31, wave = tid>>5;
  __shared__ __align__(128) half_t sIn[18*18*32];   // 20.7 KB halo tile (NHWC)
  __shared__ __align__(128) half_t sAct[16*16*64];  // 32 KB post-activation staging
  __shared__ float sSh[64];
  if(tid<64) sSh[tid]=shift[tid];
  int oy0 = PY*16, ox0 = PX*16;
  for(int i=tid;i<18*18*4;i+=256){
    int pix=i>>2, c8=i&3;
    int hy=pix/18, hx=pix%18;
    int iy=oy0+hy-1, ix=ox0+hx-1;
    v8h v = vz8h();
    if((unsigned)iy<112u && (unsigned)ix<112u)
      v = *(const v8h*)(in + ((((size_t)(b*112+iy))*112 + ix)*32) + c8*8);
    *(v8h*)(sIn + pix*32 + c8*8) = v;
  }
  __syncthreads();
  const int n0 = lane&15, k0 = (lane&16)?16:0, mb = (lane&16)?8:0;
  for(int t8=0;t8<8;++t8){
    int tile = wave + t8*8;        // 64 tiles: mt(4) x rt(16)
    int mt = tile>>4, rt = tile&15;
    v8f acc = vz8f();
#pragma unroll
    for(int tap=0;tap<9;++tap){
      int dy=tap/3, dx=tap%3;
      v16h bf = *(const v16h*)(sIn + ((rt+dy)*18 + (n0+dx))*32 + k0);
      v16h af = *(const v16h*)(wf + ((mt*9+tap)<<9) + lane*16);
      acc = wmma16(af, bf, acc);
    }
#pragma unroll
    for(int r=0;r<8;++r){
      int co = mt*16 + mb + r;
      sAct[(rt*16+n0)*64 + co] = (half_t)silu_f(acc[r] + sSh[co]);
    }
  }
  __syncthreads();
  for(int i=tid;i<4096;i+=256){
    int co=i&63, p=i>>6, oy=p>>3, ox=p&7;
    float m0 = fmaxf((float)sAct[((2*oy  )*16 + 2*ox)*64+co], (float)sAct[((2*oy  )*16 + 2*ox+1)*64+co]);
    float m1 = fmaxf((float)sAct[((2*oy+1)*16 + 2*ox)*64+co], (float)sAct[((2*oy+1)*16 + 2*ox+1)*64+co]);
    out[(((size_t)(b*56 + PY*8+oy))*56 + (PX*8+ox))*64 + co] = (half_t)fmaxf(m0,m1);
  }
}

// ---------------------------------------------------------------- conv3: implicit GEMM, WMMA f16, + global-avg (sum) reduce
__global__ __launch_bounds__(256) void k_conv3(const half_t* __restrict__ in, const half_t* __restrict__ wf,
                                               const float* __restrict__ shift, float* __restrict__ gfeat){
  int b = blockIdx.y, patch = blockIdx.x;
  int PY = patch/7, PX = patch%7;
  int tid = threadIdx.x, lane = tid&31, wave = tid>>5;
  __shared__ __align__(128) half_t sIn[10*10*64];   // 12.8 KB halo tile
  __shared__ float sFeat[128];
  __shared__ float sSh[128];
  if(tid<128){ sSh[tid]=shift[tid]; sFeat[tid]=0.f; }
  int OY0 = PY*8, OX0 = PX*8;
  for(int i=tid;i<10*10*8;i+=256){
    int pix=i>>3, c8=i&7;
    int hy=pix/10, hx=pix%10;
    int iy=OY0+hy-1, ix=OX0+hx-1;
    v8h v = vz8h();
    if((unsigned)iy<56u && (unsigned)ix<56u)
      v = *(const v8h*)(in + ((((size_t)(b*56+iy))*56 + ix)*64) + c8*8);
    *(v8h*)(sIn + pix*64 + c8*8) = v;
  }
  __syncthreads();
  const int n0 = lane&15, k0 = (lane&16)?16:0, mb = (lane&16)?8:0;
  int row = ((n0)>>3), col = n0&7;   // within 16-pixel N tile: 2 rows of 8
  for(int t4=0;t4<4;++t4){
    int tile = wave + t4*8;          // 32 tiles: nt(4) x mt(8)
    int nt = tile>>3, mt = tile&7;
    int prow = nt*2 + row;
    v8f acc = vz8f();
#pragma unroll
    for(int tap=0;tap<9;++tap){
      int dy=tap/3, dx=tap%3;
#pragma unroll
      for(int kc=0;kc<2;++kc){
        v16h bf = *(const v16h*)(sIn + ((prow+dy)*10 + (col+dx))*64 + kc*32 + k0);
        v16h af = *(const v16h*)(wf + (((mt*9+tap)*2 + kc)<<9) + lane*16);
        acc = wmma16(af, bf, acc);
      }
    }
#pragma unroll
    for(int r=0;r<8;++r){
      int co = mt*16 + mb + r;
      atomicAdd(&sFeat[co], silu_f(acc[r] + sSh[co]));
    }
  }
  __syncthreads();
  if(tid<128) atomicAdd(&gfeat[b*128+tid], sFeat[tid]);
}

// ---------------------------------------------------------------- radial MLP: 2 layers, WMMA f16 (M=batch rows)
__global__ __launch_bounds__(32) void k_mlp(const half_t* __restrict__ rad, const half_t* __restrict__ l1f,
                                            const float* __restrict__ sh1d, const half_t* __restrict__ l2f,
                                            const float* __restrict__ l2b, half_t* __restrict__ feat1d){
  int bb = blockIdx.x, lane = threadIdx.x;
  __shared__ __align__(64) half_t sG[16*64];
  const int n0 = lane&15, mb = (lane&16)?8:0, koff = (lane&16)?8:0;
  v8f acc[4]; for(int nt=0;nt<4;++nt) acc[nt]=vz8f();
  int rowg = bb*16 + n0;
#pragma unroll
  for(int kc=0;kc<4;++kc){
    const half_t* pr = rad + (size_t)rowg*128 + kc*32 + koff;
    v16h a = cat8(*(const v8h*)pr, *(const v8h*)(pr+16));
#pragma unroll
    for(int nt=0;nt<4;++nt){
      v16h bf = *(const v16h*)(l1f + ((kc*4+nt)<<9) + lane*16);
      acc[nt] = wmma16(a, bf, acc[nt]);
    }
  }
#pragma unroll
  for(int nt=0;nt<4;++nt)
#pragma unroll
    for(int r=0;r<8;++r){
      int ft = nt*16+n0;
      sG[(r+mb)*64 + ft] = (half_t)silu_f(acc[nt][r] + sh1d[ft]);
    }
  __syncthreads();
  v8f a2[4]; for(int nt=0;nt<4;++nt) a2[nt]=vz8f();
#pragma unroll
  for(int kc=0;kc<2;++kc){
    const half_t* pg = sG + n0*64 + kc*32 + koff;
    v16h a = cat8(*(const v8h*)pg, *(const v8h*)(pg+16));
#pragma unroll
    for(int nt=0;nt<4;++nt){
      v16h bf = *(const v16h*)(l2f + ((kc*4+nt)<<9) + lane*16);
      a2[nt] = wmma16(a, bf, a2[nt]);
    }
  }
#pragma unroll
  for(int nt=0;nt<4;++nt)
#pragma unroll
    for(int r=0;r<8;++r){
      int ft = nt*16+n0;
      feat1d[(size_t)(bb*16 + r + mb)*64 + ft] = (half_t)silu_f(a2[nt][r] + l2b[ft]);
    }
}

// ---------------------------------------------------------------- head: concat(feat2d/3136, feat1d) -> f1 (WMMA) -> f2
__global__ __launch_bounds__(32) void k_head(const float* __restrict__ gfeat, const half_t* __restrict__ feat1d,
                                             const half_t* __restrict__ f1f, const float* __restrict__ f1b,
                                             const float* __restrict__ f2w, const float* __restrict__ f2b,
                                             float* __restrict__ outp){
  int bb = blockIdx.x, lane = threadIdx.x;
  __shared__ float sZ[16*64];
  const int n0 = lane&15, mb = (lane&16)?8:0, koff = (lane&16)?8:0;
  const float inv_hw = 1.f/3136.f;
  int rowg = bb*16 + n0;
  v8f acc[4]; for(int nt=0;nt<4;++nt) acc[nt]=vz8f();
  for(int kc=0;kc<6;++kc){
    v16h a;
    for(int e=0;e<16;++e){
      int k = kc*32 + ((e&8)?16:0) + koff + (e&7);
      float v = (k<128) ? gfeat[rowg*128 + k]*inv_hw
                        : (float)feat1d[(size_t)rowg*64 + (k-128)];
      a[e] = (half_t)v;
    }
#pragma unroll
    for(int nt=0;nt<4;++nt){
      v16h bf = *(const v16h*)(f1f + ((kc*4+nt)<<9) + lane*16);
      acc[nt] = wmma16(a, bf, acc[nt]);
    }
  }
#pragma unroll
  for(int nt=0;nt<4;++nt)
#pragma unroll
    for(int r=0;r<8;++r){
      int ft = nt*16+n0;
      sZ[(r+mb)*64 + ft] = silu_f(acc[nt][r] + f1b[ft]);
    }
  __syncthreads();
  if(lane<16){
    float s = f2b[0];
    for(int j=0;j<64;++j) s += sZ[lane*64+j]*f2w[j];
    outp[bb*16 + lane] = s;
  }
}

// ================================================================ host
extern "C" void kernel_launch(void* const* d_in, const int* in_sizes, int n_in,
                              void* d_out, int out_size, void* d_ws, size_t ws_size,
                              hipStream_t stream) {
  const float* x    = (const float*)d_in[0];
  const float* c1_w = (const float*)d_in[1];
  const float* c1_b = (const float*)d_in[2];
  const float* bn1g = (const float*)d_in[3],  *bn1b=(const float*)d_in[4],  *bn1m=(const float*)d_in[5],  *bn1v=(const float*)d_in[6];
  const float* c2_w = (const float*)d_in[7];
  const float* c2_b = (const float*)d_in[8];
  const float* bn2g = (const float*)d_in[9],  *bn2b=(const float*)d_in[10], *bn2m=(const float*)d_in[11], *bn2v=(const float*)d_in[12];
  const float* c3_w = (const float*)d_in[13];
  const float* c3_b = (const float*)d_in[14];
  const float* bn3g = (const float*)d_in[15], *bn3b=(const float*)d_in[16], *bn3m=(const float*)d_in[17], *bn3v=(const float*)d_in[18];
  const float* l1_w = (const float*)d_in[19];
  const float* l1_b = (const float*)d_in[20];
  const float* bdg  = (const float*)d_in[21], *bdb=(const float*)d_in[22],  *bdm=(const float*)d_in[23],  *bdv=(const float*)d_in[24];
  const float* l2_w = (const float*)d_in[25];
  const float* l2_b = (const float*)d_in[26];
  const float* f1_w = (const float*)d_in[27];
  const float* f1_b = (const float*)d_in[28];
  const float* f2_w = (const float*)d_in[29];
  const float* f2_b = (const float*)d_in[30];
  float* logits = (float*)d_out;

  const int B = in_sizes[0] / (3*IMG*IMG);   // 256

  char* ws = (char*)d_ws;
  size_t off = 0;
  auto al = [&](size_t n){ off = (off + 255) & ~(size_t)255; size_t o = off; off += n; return o; };

  float* gray   = (float*)(ws + al((size_t)B*NPX*4));
  float* cfrag  = (float*)(ws + al(14*56*64*4));
  float* sfrag  = (float*)(ws + al(14*56*64*4));
  float* ure    = (float*)(ws + al((size_t)B*NPX*4));
  float* uim    = (float*)(ws + al((size_t)B*NPX*4));
  float* lm     = (float*)(ws + al((size_t)B*NPX*4));
  float* stats  = (float*)(ws + al((size_t)B*2*4));
  float* rcnt   = (float*)(ws + al(113*4));
  float* rsum   = (float*)(ws + al((size_t)B*113*4));
  half_t* rad   = (half_t*)(ws + al((size_t)B*128*2));
  float* sc1    = (float*)(ws + al(32*4));
  float* sh1    = (float*)(ws + al(32*4));
  float* sc2    = (float*)(ws + al(64*4));
  float* sh2    = (float*)(ws + al(64*4));
  float* sc3    = (float*)(ws + al(128*4));
  float* sh3    = (float*)(ws + al(128*4));
  float* sc1d   = (float*)(ws + al(64*4));
  float* sh1d   = (float*)(ws + al(64*4));
  half_t* w2f   = (half_t*)(ws + al(36*512*2));
  half_t* w3f   = (half_t*)(ws + al(144*512*2));
  half_t* o1    = (half_t*)(ws + al((size_t)B*112*112*32*2));
  half_t* o2    = (half_t*)(ws + al((size_t)B*56*56*64*2));
  float* gfeat  = (float*)(ws + al((size_t)B*128*4));
  half_t* l1f   = (half_t*)(ws + al(16*512*2));
  half_t* l2f   = (half_t*)(ws + al(8*512*2));
  half_t* f1f   = (half_t*)(ws + al(24*512*2));
  half_t* f1d   = (half_t*)(ws + al((size_t)B*64*2));
  (void)ws_size; (void)n_in; (void)out_size;

  // ---- prep: zero accumulators, fold BN, repack weights, DFT fragments
  k_zero<<<dim3((B*128+255)/256), 256, 0, stream>>>(gfeat, B*128);
  k_build_dft<<<dim3(196), 256, 0, stream>>>(cfrag, sfrag);
  k_fold<<<1, 256, 0, stream>>>(bn1g, bn1b, bn1m, bn1v, c1_b, sc1, sh1, 32);
  k_fold<<<1, 256, 0, stream>>>(bn2g, bn2b, bn2m, bn2v, c2_b, sc2, sh2, 64);
  k_fold<<<1, 256, 0, stream>>>(bn3g, bn3b, bn3m, bn3v, c3_b, sc3, sh3, 128);
  k_fold<<<1, 256, 0, stream>>>(bdg, bdb, bdm, bdv, l1_b, sc1d, sh1d, 64);
  k_pack_wconv<<<dim3(72), 256, 0, stream>>>(c2_w, sc2, w2f, 64, 32);
  k_pack_wconv<<<dim3(288), 256, 0, stream>>>(c3_w, sc3, w3f, 128, 64);
  k_pack_lin<<<dim3(16), 256, 0, stream>>>(l1_w, sc1d, l1f, 64, 112, 128);
  k_pack_lin<<<dim3(8),  256, 0, stream>>>(l2_w, nullptr, l2f, 64, 64, 64);
  k_pack_lin<<<dim3(24), 256, 0, stream>>>(f1_w, nullptr, f1f, 64, 192, 192);

  // ---- FFT features
  k_gray<<<dim3(196, B), 256, 0, stream>>>(x, gray);
  k_dft1<<<dim3(196, B), 32, 0, stream>>>(gray, cfrag, sfrag, ure, uim);
  k_dft2<<<dim3(196, B), 32, 0, stream>>>(ure, uim, cfrag, sfrag, lm);
  k_stats2d<<<dim3(B), 256, 0, stream>>>(lm, stats);
  k_radial_counts<<<dim3(1), 256, 0, stream>>>(rcnt);
  k_radial_sums<<<dim3(B), 256, 0, stream>>>(lm, rsum);
  k_radial_norm<<<dim3(B), 128, 0, stream>>>(rsum, rcnt, rad);

  // ---- CNN path
  k_conv1<<<dim3(49, B), 256, 0, stream>>>(lm, stats, c1_w, sc1, sh1, o1);
  k_conv2<<<dim3(49, B), 256, 0, stream>>>(o1, w2f, sh2, o2);
  k_conv3<<<dim3(49, B), 256, 0, stream>>>(o2, w3f, sh3, gfeat);

  // ---- MLP path + head
  k_mlp<<<dim3(B/16), 32, 0, stream>>>(rad, l1f, sh1d, l2f, l2_b, f1d);
  k_head<<<dim3(B/16), 32, 0, stream>>>(gfeat, f1d, f1f, f1_b, f2_w, f2_b, logits);
}
